// OctreeCaps_60129542144594
// MI455X (gfx1250) — compile-verified, matrix operand order checked by reading, not run on hardware
//
#include <hip/hip_runtime.h>
#include <hip/hip_bf16.h>

// ---------------- problem constants ----------------
#define B_   32      // batch
#define CCH  32      // channels
#define P_   1287    // prim caps
#define L1_  937
#define L2_  234
#define L3_  116
#define LC   40      // latent caps
#define LV   40      // latent vec
#define LIN_ 4681
#define EPSB 1e-5f
#define NSLOPE 0.2f

typedef __attribute__((ext_vector_type(16))) __bf16 v16bf;
typedef __attribute__((ext_vector_type(8)))  __bf16 bf16x8;
typedef __attribute__((ext_vector_type(2)))  __bf16 bf16x2;
typedef __attribute__((ext_vector_type(8)))  float  v8f;

// native f32->bf16 (RNE) so clang emits v_cvt(_pk)_bf16_f32 instead of int ALU
__device__ __forceinline__ unsigned short f2bfbits(float f) {
  return __builtin_bit_cast(unsigned short, (__bf16)f);
}
__device__ __forceinline__ unsigned pack2bf(float lo, float hi) {
  bf16x2 t; t[0] = (__bf16)lo; t[1] = (__bf16)hi;
  return __builtin_bit_cast(unsigned, t);
}
__device__ __forceinline__ float bf2f(unsigned short h) {
  return __builtin_bit_cast(float, (unsigned)h << 16);
}
__device__ __forceinline__ float lrelu(float y) { return y >= 0.f ? y : NSLOPE * y; }

// ---------------- conv front-end (latency-class work, plain VALU) ----------------
__global__ __launch_bounds__(256) void conv1_bn_k(
    const float* __restrict__ data, const float* __restrict__ w,
    const float* __restrict__ g, const float* __restrict__ be,
    const float* __restrict__ mn, const float* __restrict__ vr,
    float* __restrict__ x1) {
  int tid = blockIdx.x * 256 + threadIdx.x;
  if (tid >= B_ * CCH * L1_) return;
  int i = tid % L1_; int t2 = tid / L1_; int co = t2 % CCH; int b = t2 / CCH;
  const float* in = data + (size_t)b * LIN_;
  float acc = 0.f;
#pragma unroll
  for (int k = 0; k < 5; ++k) {
    int ix = 5 * i + k - 2;
    float xv = (ix >= 0 && ix < LIN_) ? in[ix] : 0.f;
    acc += w[co * 5 + k] * xv;
  }
  float sc = g[co] * rsqrtf(vr[co] + EPSB);
  x1[tid] = lrelu((acc - mn[co]) * sc + be[co]);
}

__global__ __launch_bounds__(256) void conv2_bn_k(
    const float* __restrict__ x1, const float* __restrict__ w,
    const float* __restrict__ g, const float* __restrict__ be,
    const float* __restrict__ mn, const float* __restrict__ vr,
    float* __restrict__ x2) {
  int tid = blockIdx.x * 256 + threadIdx.x;
  if (tid >= B_ * CCH * L2_) return;
  int j = tid % L2_; int t2 = tid / L2_; int co = t2 % CCH; int b = t2 / CCH;
  float acc = 0.f;
  for (int ci = 0; ci < CCH; ++ci) {
    const float* r  = x1 + ((size_t)b * CCH + ci) * L1_ + 4 * j;
    const float* wr = w + (co * CCH + ci) * 5;
#pragma unroll
    for (int k = 0; k < 5; ++k) acc += wr[k] * r[k];
  }
  float sc = g[co] * rsqrtf(vr[co] + EPSB);
  x2[tid] = lrelu((acc - mn[co]) * sc + be[co]);
}

__global__ __launch_bounds__(256) void conv3_bn_k(
    const float* __restrict__ x2, const float* __restrict__ w,
    const float* __restrict__ g, const float* __restrict__ be,
    const float* __restrict__ mn, const float* __restrict__ vr,
    float* __restrict__ x3) {
  int tid = blockIdx.x * 256 + threadIdx.x;
  if (tid >= B_ * CCH * L3_) return;
  int t = tid % L3_; int t2 = tid / L3_; int co = t2 % CCH; int b = t2 / CCH;
  float acc = 0.f;
  for (int ci = 0; ci < CCH; ++ci) {
    const float* r  = x2 + ((size_t)b * CCH + ci) * L2_ + 2 * t;
    const float* wr = w + (co * CCH + ci) * 4;
#pragma unroll
    for (int k = 0; k < 4; ++k) acc += wr[k] * r[k];
  }
  float sc = g[co] * rsqrtf(vr[co] + EPSB);
  x3[tid] = lrelu((acc - mn[co]) * sc + be[co]);
}

// 1x1 conv over concat + BN + LReLU + transpose + bf16 pack as xb[p][b][s]
__global__ __launch_bounds__(256) void lin_bn_pack_k(
    const float* __restrict__ x1, const float* __restrict__ x2, const float* __restrict__ x3,
    const float* __restrict__ lw, const float* __restrict__ g, const float* __restrict__ be,
    const float* __restrict__ mn, const float* __restrict__ vr,
    unsigned short* __restrict__ xb) {
  int tid = blockIdx.x * 256 + threadIdx.x;
  if (tid >= B_ * P_) return;
  int b = tid / P_, p = tid - b * P_;
  float xin[CCH];
#pragma unroll 4
  for (int ci = 0; ci < CCH; ++ci) {
    float xv;
    if (p < L1_)              xv = x1[((size_t)b * CCH + ci) * L1_ + p];
    else if (p < L1_ + L2_)   xv = x2[((size_t)b * CCH + ci) * L2_ + (p - L1_)];
    else                      xv = x3[((size_t)b * CCH + ci) * L3_ + (p - L1_ - L2_)];
    xin[ci] = xv;
  }
  unsigned short* o = xb + ((size_t)p * B_ + b) * CCH;
  for (int co = 0; co < CCH; ++co) {
    float acc = 0.f;
#pragma unroll 8
    for (int ci = 0; ci < CCH; ++ci) acc += lw[co * CCH + ci] * xin[ci];
    float sc = g[co] * rsqrtf(vr[co] + EPSB);
    o[co] = f2bfbits(lrelu((acc - mn[co]) * sc + be[co]));
  }
}

// ---------------- capsule einsum via V_WMMA_F32_16X16X32_BF16 ----------------
// Per p: A = W[l,p,v,s] as [1600 x 32] (row m=l*40+v, K=s), converted to bf16
// in regs; B = x[b,p,s] as [32(K) x 32(N=b)]. One wave owns one 16-row m-tile
// and BOTH n-tiles (A loaded once -> W read exactly once = 263MB roofline).
// p processed in PAIRS so accumulators are stored as packed-bf16 dwords.
// Output u_hat bf16 in [B][L][V][P] (132MB -> resident in 192MB L2 for routing).
#define PCHUNK 8
#define NCHUNK 161   // ceil(1287/8)

__device__ __forceinline__ v16bf load_a_bf16(const float* __restrict__ Arow, int kb) {
  float4 x0 = *(const float4*)(Arow + kb);
  float4 x1 = *(const float4*)(Arow + kb + 4);
  float4 x2 = *(const float4*)(Arow + kb + 16);
  float4 x3 = *(const float4*)(Arow + kb + 20);
  v16bf a;
  a[0]  = (__bf16)x0.x; a[1]  = (__bf16)x0.y; a[2]  = (__bf16)x0.z; a[3]  = (__bf16)x0.w;
  a[4]  = (__bf16)x1.x; a[5]  = (__bf16)x1.y; a[6]  = (__bf16)x1.z; a[7]  = (__bf16)x1.w;
  a[8]  = (__bf16)x2.x; a[9]  = (__bf16)x2.y; a[10] = (__bf16)x2.z; a[11] = (__bf16)x2.w;
  a[12] = (__bf16)x3.x; a[13] = (__bf16)x3.y; a[14] = (__bf16)x3.z; a[15] = (__bf16)x3.w;
  return a;
}

__device__ __forceinline__ v16bf load_b_bf16(const unsigned short* __restrict__ base) {
  bf16x8 lo = *(const bf16x8*)(base);        // K kb..kb+7
  bf16x8 hi = *(const bf16x8*)(base + 16);   // K kb+16..kb+23
  v16bf b;
#pragma unroll
  for (int i = 0; i < 8; ++i) { b[i] = lo[i]; b[i + 8] = hi[i]; }
  return b;
}

__global__ __launch_bounds__(256) void caps_gemm_wmma_k(
    const float* __restrict__ W, const unsigned short* __restrict__ xb,
    unsigned short* __restrict__ uh) {
  const int wave = threadIdx.x >> 5, lane = threadIdx.x & 31;
  int t = blockIdx.x * 8 + wave;
  if (t >= 100 * NCHUNK) return;                  // wave-uniform: EXEC stays all-1s
  const int mt   = t / NCHUNK;
  const int p0   = (t % NCHUNK) * PCHUNK;         // even (multiple of 8)
  const int half = lane >> 4;                     // K-half select per ISA layout
  const int lidx = lane & 15;
  const int mrow = mt * 16 + lidx;
  const int l = mrow / LV, v = mrow - l * LV;
  const int kb = half * 8;                        // lanes16-31 hold K 8-15 / 24-31
  const float* Abase = W + ((size_t)l * P_ * LV + v) * CCH;

  for (int pi = 0; pi < PCHUNK; pi += 2) {
    int p = p0 + pi;
    if (p >= P_) break;                           // uniform per wave
    const bool pairv = (p + 1 < P_);
    const int p1 = pairv ? p + 1 : p;             // keep loads in-bounds on tail

    v16bf aA = load_a_bf16(Abase + (size_t)p  * LV * CCH, kb);
    v16bf aB = load_a_bf16(Abase + (size_t)p1 * LV * CCH, kb);

    const unsigned short* BpA = xb + (size_t)p  * (B_ * CCH);
    const unsigned short* BpB = xb + (size_t)p1 * (B_ * CCH);
    v16bf b0A = load_b_bf16(BpA + lidx * CCH + kb);
    v16bf b1A = load_b_bf16(BpA + (16 + lidx) * CCH + kb);
    v16bf b0B = load_b_bf16(BpB + lidx * CCH + kb);
    v16bf b1B = load_b_bf16(BpB + (16 + lidx) * CCH + kb);

    v8f c0A = {}, c1A = {}, c0B = {}, c1B = {};
    c0A = __builtin_amdgcn_wmma_f32_16x16x32_bf16(false, aA, false, b0A, (short)0, c0A, false, false);
    c1A = __builtin_amdgcn_wmma_f32_16x16x32_bf16(false, aA, false, b1A, (short)0, c1A, false, false);
    c0B = __builtin_amdgcn_wmma_f32_16x16x32_bf16(false, aB, false, b0B, (short)0, c0B, false, false);
    c1B = __builtin_amdgcn_wmma_f32_16x16x32_bf16(false, aB, false, b1B, (short)0, c1B, false, false);

    // D layout: VGPR r -> M = mt*16 + half*8 + r ; N = lane&15 (+16 for tile 1).
    // Consecutive p are adjacent in memory -> packed bf16x2 dword stores.
#pragma unroll
    for (int r = 0; r < 8; ++r) {
      int mm = mt * 16 + half * 8 + r;
      size_t i0 = ((size_t)lidx        * (LC * LV) + mm) * P_ + p;
      size_t i1 = ((size_t)(lidx + 16) * (LC * LV) + mm) * P_ + p;
      if (pairv) {                                // p even; P_ odd, so 4B aligned only for even mm
        *(unsigned*)(uh + i0) = pack2bf(c0A[r], c0B[r]);
        *(unsigned*)(uh + i1) = pack2bf(c1A[r], c1B[r]);
      } else {
        uh[i0] = f2bfbits(c0A[r]);
        uh[i1] = f2bfbits(c1A[r]);
      }
    }
  }
}

// ---------------- dynamic routing (u_hat bf16, L2-resident) ----------------
__global__ __launch_bounds__(256) void zero_f32_k(float* p, int n) {
  int i = blockIdx.x * 256 + threadIdx.x;
  if (i < n) p[i] = 0.f;
}

// softmax over l (axis=1) per (b,p)
__global__ __launch_bounds__(256) void route_softmax_k(
    const float* __restrict__ bij, float* __restrict__ c) {
  int tid = blockIdx.x * 256 + threadIdx.x;
  if (tid >= B_ * P_) return;
  int b = tid / P_, p = tid - b * P_;
  const float* br = bij + (size_t)b * LC * P_ + p;
  float* cr = c + (size_t)b * LC * P_ + p;
  float mx = -1e30f;
  for (int l = 0; l < LC; ++l) mx = fmaxf(mx, br[(size_t)l * P_]);
  float s = 0.f;
  for (int l = 0; l < LC; ++l) { float e = __expf(br[(size_t)l * P_] - mx); cr[(size_t)l * P_] = e; s += e; }
  float inv = 1.f / s;
  for (int l = 0; l < LC; ++l) cr[(size_t)l * P_] *= inv;
}

// s[v] = sum_p c*uh ; squash -> vvec[b][l][v].  One block per (b,l).
__global__ __launch_bounds__(256) void route_sv_k(
    const float* __restrict__ c, const unsigned short* __restrict__ uh,
    float* __restrict__ vvec) {
  __shared__ float cbuf[P_];
  __shared__ float wred[8];
  __shared__ float sv[LV];
  __shared__ float sfac;
  int bl = blockIdx.x;                 // b*LC + l
  int tid = threadIdx.x, lane = tid & 31, wv = tid >> 5;
  const float* cr = c + (size_t)bl * P_;
  for (int p = tid; p < P_; p += 256) cbuf[p] = cr[p];
  __syncthreads();
  const unsigned short* ub = uh + (size_t)bl * (LV * P_);
  for (int v = 0; v < LV; ++v) {
    const unsigned short* ur = ub + (size_t)v * P_;
    float acc = 0.f;
    for (int p = tid; p < P_; p += 256) acc += cbuf[p] * bf2f(ur[p]);
#pragma unroll
    for (int off = 16; off > 0; off >>= 1) acc += __shfl_xor(acc, off, 32);
    if (lane == 0) wred[wv] = acc;
    __syncthreads();
    if (tid == 0) {
      float s = 0.f;
      for (int w = 0; w < 8; ++w) s += wred[w];
      sv[v] = s;
    }
    __syncthreads();
  }
  if (tid == 0) {
    float sn = 0.f;
    for (int v = 0; v < LV; ++v) sn += sv[v] * sv[v];
    sfac = sn / ((1.f + sn) * sqrtf(sn));
  }
  __syncthreads();
  if (tid < LV) vvec[(size_t)bl * LV + tid] = sv[tid] * sfac;
}

// b_ij[b,l,p] += sum_v vvec[b,l,v]*uh[b,l,v,p].  One block per (b,l).
__global__ __launch_bounds__(256) void route_bupd_k(
    const float* __restrict__ vvec, const unsigned short* __restrict__ uh,
    float* __restrict__ bij) {
  __shared__ float vv[LV];
  int bl = blockIdx.x, tid = threadIdx.x;
  if (tid < LV) vv[tid] = vvec[(size_t)bl * LV + tid];
  __syncthreads();
  const unsigned short* ub = uh + (size_t)bl * (LV * P_);
  float* br = bij + (size_t)bl * P_;
  for (int p = tid; p < P_; p += 256) {
    float acc = 0.f;
#pragma unroll 8
    for (int v = 0; v < LV; ++v) acc += vv[v] * bf2f(ub[(size_t)v * P_ + p]);
    br[p] += acc;
  }
}

__global__ __launch_bounds__(256) void route_out_k(
    const float* __restrict__ vvec, float* __restrict__ out) {
  int i = blockIdx.x * 256 + threadIdx.x;
  if (i >= B_ * LC) return;
  float s = 0.f;
  for (int v = 0; v < LV; ++v) s += vvec[(size_t)i * LV + v];
  out[i] = s;
}

// ---------------- launch ----------------
extern "C" void kernel_launch(void* const* d_in, const int* in_sizes, int n_in,
                              void* d_out, int out_size, void* d_ws, size_t ws_size,
                              hipStream_t stream) {
  (void)in_sizes; (void)n_in; (void)out_size; (void)ws_size;
  const float* data = (const float*)d_in[0];
  const float* w1   = (const float*)d_in[1];
  const float* w2   = (const float*)d_in[2];
  const float* w3   = (const float*)d_in[3];
  const float* lw   = (const float*)d_in[4];
  const float* bg   = (const float*)d_in[5];   // [4][32]
  const float* bb   = (const float*)d_in[6];
  const float* bm   = (const float*)d_in[7];
  const float* bv   = (const float*)d_in[8];
  const float* Wc   = (const float*)d_in[9];   // [40][1287][40][32]
  float* out = (float*)d_out;

  // workspace carve-up (all sizes are 256B multiples)
  char* ws = (char*)d_ws;
  const size_t SZ_X1 = (size_t)B_ * CCH * L1_ * 4;       // 3,837,952
  const size_t SZ_X2 = (size_t)B_ * CCH * L2_ * 4;       //   958,464
  const size_t SZ_X3 = (size_t)B_ * CCH * L3_ * 4;       //   475,136
  const size_t SZ_XB = (size_t)P_ * B_ * CCH * 2;        // 2,635,776
  const size_t SZ_UH = (size_t)B_ * LC * LV * P_ * 2;    // 131,788,800 (bf16, fits 192MB L2)
  const size_t SZ_BJ = (size_t)B_ * LC * P_ * 4;         // 6,589,440
  float*          x1   = (float*)(ws);
  float*          x2   = (float*)(ws + SZ_X1);
  float*          x3   = (float*)(ws + SZ_X1 + SZ_X2);
  unsigned short* xb   = (unsigned short*)(ws + SZ_X1 + SZ_X2 + SZ_X3);
  unsigned short* uh   = (unsigned short*)(ws + SZ_X1 + SZ_X2 + SZ_X3 + SZ_XB);
  float*          bij  = (float*)(ws + SZ_X1 + SZ_X2 + SZ_X3 + SZ_XB + SZ_UH);
  float*          cij  = (float*)(ws + SZ_X1 + SZ_X2 + SZ_X3 + SZ_XB + SZ_UH + SZ_BJ);
  float*          vvec = (float*)(ws + SZ_X1 + SZ_X2 + SZ_X3 + SZ_XB + SZ_UH + 2 * SZ_BJ);

  const int NBIJ = B_ * LC * P_;
  zero_f32_k<<<(NBIJ + 255) / 256, 256, 0, stream>>>(bij, NBIJ);

  conv1_bn_k<<<(B_ * CCH * L1_ + 255) / 256, 256, 0, stream>>>(
      data, w1, bg + 0, bb + 0, bm + 0, bv + 0, x1);
  conv2_bn_k<<<(B_ * CCH * L2_ + 255) / 256, 256, 0, stream>>>(
      x1, w2, bg + 32, bb + 32, bm + 32, bv + 32, x2);
  conv3_bn_k<<<(B_ * CCH * L3_ + 255) / 256, 256, 0, stream>>>(
      x2, w3, bg + 64, bb + 64, bm + 64, bv + 64, x3);
  lin_bn_pack_k<<<(B_ * P_ + 255) / 256, 256, 0, stream>>>(
      x1, x2, x3, lw, bg + 96, bb + 96, bm + 96, bv + 96, xb);

  // 100 m-tiles * 161 p-chunks = 16100 waves, 8 waves/block
  caps_gemm_wmma_k<<<(100 * NCHUNK + 7) / 8, 256, 0, stream>>>(Wc, xb, uh);

  for (int it = 0; it < 3; ++it) {
    route_softmax_k<<<(B_ * P_ + 255) / 256, 256, 0, stream>>>(bij, cij);
    route_sv_k<<<B_ * LC, 256, 0, stream>>>(cij, uh, vvec);
    if (it < 2)
      route_bupd_k<<<B_ * LC, 256, 0, stream>>>(vvec, uh, bij);
  }
  route_out_k<<<(B_ * LC + 255) / 256, 256, 0, stream>>>(vvec, out);
}
